// SkipACell_61890478735974
// MI455X (gfx1250) — compile-verified, matrix operand order checked by reading, not run on hardware
//
#include <hip/hip_runtime.h>

// ---------------------------------------------------------------------------
// SkipLSTM (SkipACell) for MI455X / gfx1250, wave32, WMMA bf16.
//   T=512, B=64, I=1024, H=1024, gates = 4H = 4096, K = I+H = 2048.
//
// v3: persistent kernel, now with split-K across waves for occupancy:
//   - 64 blocks (one per 16 hidden columns) x 1024 threads = 32 waves
//   - wave = (kh, g, mt): kh=0 computes the x-half of K, kh=1 the h-half.
//     2 waves/SIMD -> the L2-warm x-half hides the barrier-fresh h-half
//     latency; per-wave WMMA chain halves to 32 (shorter serial path).
//   - partial sums combined from two LDS slabs (32 KB) in elementwise phase
//   - non-temporal stores for the 128 MB streaming `out` so the 16 MB bf16
//     weight matrix + x stay resident in the 192 MB L2 across all steps.
// Device-wide generation barrier (monotonic counter + fences) between steps;
// h state ping-pongs between two bf16 buffers so one barrier/step suffices.
// ---------------------------------------------------------------------------

typedef __attribute__((ext_vector_type(16))) __bf16 v16bf;
typedef __attribute__((ext_vector_type(8)))  __bf16 v8bf;
typedef __attribute__((ext_vector_type(8)))  float  v8f;

#define TT 512
#define BB 64
#define II 1024
#define HH 1024
#define KK 2048   // I + H
#define GG 4096   // 4*H

// ---- workspace layout (bytes) ----
#define WS_WCAT 0                           // 4096*2048*2      = 16,777,216
#define WS_XBF  (16777216)                  // 512*64*1024*2    = 67,108,864
#define WS_HBF0 (WS_XBF  + 67108864)        // 64*1024*2        = 131,072
#define WS_HBF1 (WS_HBF0 + 131072)          // 64*1024*2        = 131,072
#define WS_C    (WS_HBF1 + 131072)          // 64*1024*4        = 262,144
#define WS_SC   (WS_C    + 262144)          // 64*1024*4        = 262,144
#define WS_BV   (WS_SC   + 262144)          // 4096*4           = 16,384
#define WS_BAR  (WS_BV   + 16384)           // barrier counter  = 128
// total ~84.7 MB

__device__ __forceinline__ v16bf load_frag(const __bf16* p) {
    // CDNA5 16-bit A/B fragment: this lane's 16 values are K=[kbase..kbase+7]
    // and K=[kbase+16..kbase+23] -> two contiguous 16-byte chunks.
    v8bf lo = *(const v8bf*)(p);
    v8bf hi = *(const v8bf*)(p + 16);
    v16bf r;
#pragma unroll
    for (int i = 0; i < 8; ++i) { r[i] = lo[i]; r[i + 8] = hi[i]; }
    return r;
}

__device__ __forceinline__ float sigmoidf_(float x) {
    return 1.0f / (1.0f + __expf(-x));
}

// ---------------------------------------------------------------------------
// Prep kernels
// ---------------------------------------------------------------------------
__global__ void prep_weights(const float* __restrict__ w_ih,
                             const float* __restrict__ w_hh,
                             const float* __restrict__ b_ih,
                             const float* __restrict__ b_hh,
                             const int*   __restrict__ bias_flag,
                             __bf16* __restrict__ Wcat,
                             float*  __restrict__ bvec) {
    const int stride = gridDim.x * blockDim.x;
    const long long total = (long long)GG * KK;
    for (long long i = blockIdx.x * blockDim.x + threadIdx.x; i < total; i += stride) {
        int r = (int)(i >> 11);        // row in [0,4096)
        int k = (int)(i & 2047);       // col in [0,2048)
        float v = (k < II) ? w_ih[(long long)r * II + k]
                           : w_hh[(long long)r * HH + (k - II)];
        Wcat[i] = (__bf16)v;
    }
    const int bf = bias_flag[0];
    for (int i = blockIdx.x * blockDim.x + threadIdx.x; i < GG; i += stride) {
        bvec[i] = bf ? (b_ih[i] + b_hh[i]) : 0.0f;
    }
}

__global__ void prep_x(const float* __restrict__ x, __bf16* __restrict__ x_bf) {
    const long long total = (long long)TT * BB * II;
    const long long stride = (long long)gridDim.x * blockDim.x;
    for (long long i = blockIdx.x * blockDim.x + threadIdx.x; i < total; i += stride) {
        x_bf[i] = (__bf16)x[i];
    }
}

__global__ void prep_state(const float* __restrict__ hidden,
                           const float* __restrict__ candidate,
                           __bf16* __restrict__ h_bf0,
                           float*  __restrict__ c,
                           float*  __restrict__ sc,
                           int*    __restrict__ bar) {
    const int total = BB * HH;
    const int stride = gridDim.x * blockDim.x;
    const int gid = blockIdx.x * blockDim.x + threadIdx.x;
    for (int i = gid; i < total; i += stride) {
        float h = hidden[i];
        h_bf0[i] = (__bf16)h;       // step 0 reads buffer 0
        sc[i]    = h;               // initial skip connection = hidden
        c[i]     = candidate[i];    // cell state
    }
    if (gid == 0) bar[0] = 0;       // reset grid barrier every call
}

// ---------------------------------------------------------------------------
// Persistent recurrence kernel.
// Grid: 64 blocks (block nb owns hidden columns [nb*16, nb*16+16)); all
// co-resident -> spin barrier is safe. Block: 1024 threads = 32 waves.
// wave = (kh, g, mt): K-half, gate section, batch-row tile.
// ---------------------------------------------------------------------------
__global__ void __launch_bounds__(1024)
skip_lstm_persistent(const __bf16* __restrict__ x_bf,   // [T][B][I]
                     const __bf16* __restrict__ Wcat,   // [4H][K] row-major
                     const float*  __restrict__ bvec,   // [4H]
                     __bf16* __restrict__ h_bf0,        // [B][H] ping
                     __bf16* __restrict__ h_bf1,        // [B][H] pong
                     float*  __restrict__ c_ws,         // [B][H]
                     float*  __restrict__ sc_ws,        // [B][H]
                     int*    __restrict__ bar,          // grid barrier counter
                     float*  __restrict__ out) {        // d_out
    __shared__ float lds[2][4][BB][16];   // 32 KB: per K-half gate tiles

    const int nb    = blockIdx.x;          // column tile of H
    const int lane  = threadIdx.x & 31;
    const int wave  = threadIdx.x >> 5;
    const int g     = wave & 3;            // gate section (i,f,c,o)
    const int mt    = (wave >> 2) & 3;     // batch-row tile
    const int kh    = wave >> 4;           // K-half: 0 = x part, 1 = h part
    const int m0    = mt * 16;
    const int l16   = lane & 15;
    const int kbase = (lane >> 4) * 8;     // per-lane K sub-offset (A/B frag)

    // A: batch row for this lane; B: W row = gate column for this lane (B^T feed)
    const int m    = m0 + l16;
    const int wrow = g * HH + nb * 16 + l16;

    const __bf16* arow_x0 = x_bf + (long long)m * II;      // +t*B*I per step
    const __bf16* brow    = Wcat + (long long)wrow * KK + kh * II;
    const int     nbars   = (int)gridDim.x;

    for (int t = 0; t < TT; ++t) {
        const __bf16* h_src = (t & 1) ? h_bf1 : h_bf0;
        __bf16*       h_dst = (t & 1) ? h_bf0 : h_bf1;

        // kh==0 waves sweep x_t (L2-warm stream), kh==1 waves sweep fresh h.
        const __bf16* arow = kh ? (h_src + (long long)m * HH)
                                : (arow_x0 + (long long)t * (BB * II));

        v8f acc = {};
#pragma unroll 4
        for (int k0 = 0; k0 < II; k0 += 32) {
            v16bf a = load_frag(arow + k0 + kbase);
            v16bf b = load_frag(brow + k0 + kbase);
            acc = __builtin_amdgcn_wmma_f32_16x16x32_bf16(false, a, false, b,
                                                          (short)0, acc, false, false);
        }

        // C/D layout: VGPR r -> M = m0 + r + (lane>=16 ? 8 : 0), N = lane&15
        const int mrow = m0 + ((lane >> 4) << 3);
#pragma unroll
        for (int r = 0; r < 8; ++r) {
            lds[kh][g][mrow + r][l16] = acc[r];
        }
        __syncthreads();

        // Elementwise LSTM update: 64x16 = 1024 elements, 1024 threads -> 1 each.
        const bool do_add = ((t & 3) == 3);
        {
            const int e  = threadIdx.x;
            const int em = e >> 4;             // batch row
            const int en = e & 15;             // column within tile
            const int ng = nb * 16 + en;       // global hidden column

            float ig = sigmoidf_(lds[0][0][em][en] + lds[1][0][em][en] + bvec[0 * HH + ng]);
            float fg = sigmoidf_(lds[0][1][em][en] + lds[1][1][em][en] + bvec[1 * HH + ng]);
            float cs = tanhf    (lds[0][2][em][en] + lds[1][2][em][en] + bvec[2 * HH + ng]);
            float og = sigmoidf_(lds[0][3][em][en] + lds[1][3][em][en] + bvec[3 * HH + ng]);

            const int sidx = em * HH + ng;
            float c_new = fg * c_ws[sidx] + ig * cs;
            float h_new = og * tanhf(c_new);
            if (do_add) {
                h_new += sc_ws[sidx];
                sc_ws[sidx] = h_new;
            }
            c_ws[sidx] = c_new;
            h_dst[sidx] = (__bf16)h_new;       // next step reads the other buffer

            // Streaming output: non-temporal so Wcat/x stay L2-resident.
            __builtin_nontemporal_store(h_new, &out[(long long)t * BB * HH + sidx]);
            if (t == TT - 1) {
                __builtin_nontemporal_store(
                    h_new, &out[(long long)TT * BB * HH + sidx]);                       // final h
                __builtin_nontemporal_store(
                    c_new, &out[(long long)TT * BB * HH + (long long)BB * HH + sidx]);  // final c
            }
        }

        // ---- device-wide generation barrier (monotonic counter) ----
        __threadfence();                 // release: h_dst visible at device scope
        __syncthreads();                 // all block stores issued & fenced
        if (threadIdx.x == 0) {
            atomicAdd(bar, 1);
            const int target = (t + 1) * nbars;
            while (__hip_atomic_load(bar, __ATOMIC_ACQUIRE,
                                     __HIP_MEMORY_SCOPE_AGENT) < target) {
                __builtin_amdgcn_s_sleep(8);
            }
        }
        __syncthreads();                 // re-converge (EXEC all-1 for next WMMA)
        __threadfence();                 // acquire: invalidate stale h lines
    }
}

// ---------------------------------------------------------------------------
extern "C" void kernel_launch(void* const* d_in, const int* in_sizes, int n_in,
                              void* d_out, int out_size, void* d_ws, size_t ws_size,
                              hipStream_t stream) {
    const float* x         = (const float*)d_in[0];
    const float* hidden    = (const float*)d_in[1];
    const float* candidate = (const float*)d_in[2];
    const float* w_ih      = (const float*)d_in[3];
    const float* w_hh      = (const float*)d_in[4];
    const float* b_ih      = (const float*)d_in[5];
    const float* b_hh      = (const float*)d_in[6];
    const int*   bias_flag = (const int*)d_in[7];
    float* out = (float*)d_out;

    char* ws = (char*)d_ws;
    __bf16* Wcat  = (__bf16*)(ws + WS_WCAT);
    __bf16* x_bf  = (__bf16*)(ws + WS_XBF);
    __bf16* h_bf0 = (__bf16*)(ws + WS_HBF0);
    __bf16* h_bf1 = (__bf16*)(ws + WS_HBF1);
    float*  c_ws  = (float*)(ws + WS_C);
    float*  sc_ws = (float*)(ws + WS_SC);
    float*  bvec  = (float*)(ws + WS_BV);
    int*    bar   = (int*)(ws + WS_BAR);

    prep_weights<<<1024, 256, 0, stream>>>(w_ih, w_hh, b_ih, b_hh, bias_flag, Wcat, bvec);
    prep_x<<<4096, 256, 0, stream>>>(x, x_bf);
    prep_state<<<256, 256, 0, stream>>>(hidden, candidate, h_bf0, c_ws, sc_ws, bar);

    skip_lstm_persistent<<<HH / 16, 1024, 0, stream>>>(x_bf, Wcat, bvec,
                                                       h_bf0, h_bf1, c_ws, sc_ws,
                                                       bar, out);
}
